// Model_77601469104701
// MI455X (gfx1250) — compile-verified, matrix operand order checked by reading, not run on hardware
//
#include <hip/hip_runtime.h>
#include <hip/hip_bf16.h>

// ---------------------------------------------------------------------------
// BiLSTM-CRF forward for MI455X (gfx1250, wave32, WMMA + TDM).
//   B=32, T=256, V=30000, E=300(pad 320), H=512, L=2, C=17(pad 32)
// All GEMMs run on v_wmma_f32_16x16x32_bf16 (bf16 in, f32 accum).
// Recurrence: persistent kernel, 16 WGs, global ping-pong h + grid barrier;
// per-step h staging into LDS via the Tensor Data Mover (tensor_load_to_lds,
// D# pad feature produces the bank-conflict-free 514-ushort row stride).
// ---------------------------------------------------------------------------

#define DEVI __device__ __forceinline__

typedef __attribute__((ext_vector_type(16))) __bf16 bf16x16;
typedef __attribute__((ext_vector_type(8)))  float  f32x8;
typedef __attribute__((ext_vector_type(4)))  unsigned int u32x4;
typedef __attribute__((ext_vector_type(8)))  int  i32x8;
typedef __attribute__((ext_vector_type(4)))  int  i32x4;

constexpr int Bq = 32, Tq = 256, Eq = 300, EP = 320, Hq = 512, Cq = 17;
constexpr int G4 = 2048;   // 4*H
constexpr int H2 = 1024;   // 2*H
constexpr int NWG = 16;    // persistent LSTM workgroups

DEVI unsigned short f2bf(float f) {           // round-to-nearest-even
  unsigned u = __float_as_uint(f);
  u += 0x7FFFu + ((u >> 16) & 1u);
  return (unsigned short)(u >> 16);
}
DEVI float bf2f(unsigned short s) { return __uint_as_float(((unsigned)s) << 16); }
DEVI float sigf(float x) { return 1.f / (1.f + __expf(-x)); }

// K-position inside a 16-bit WMMA fragment: element e (0..15), lane-half.
DEVI int kmap(int e, int half) { return e + ((e >= 8) ? 8 : 0) + 8 * half; }

union BFrag { bf16x16 v; unsigned short s[16]; uint4 q[2]; };
union U16   { uint4 q; unsigned short s[8]; };

DEVI f32x8 wmma_bf16(bf16x16 a, bf16x16 b, f32x8 c) {
  return __builtin_amdgcn_wmma_f32_16x16x32_bf16(false, a, false, b,
                                                 (short)0, c, false, false);
}

DEVI void gsync(unsigned int* cnt, unsigned int nwg, unsigned int phase) {
  __syncthreads();
  __threadfence();
  if (threadIdx.x == 0) {
    __hip_atomic_fetch_add(cnt, 1u, __ATOMIC_RELEASE, __HIP_MEMORY_SCOPE_AGENT);
    while (__hip_atomic_load(cnt, __ATOMIC_ACQUIRE, __HIP_MEMORY_SCOPE_AGENT)
           < nwg * phase) {
      __builtin_amdgcn_s_sleep(1);
    }
  }
  __syncthreads();
}

// --- Tensor DMA descriptor helpers (CDNA5 D#, ISA 08_async_tensor §8) ------
// Group 0: [1:0]=count=1, [63:32]=lds_addr, [120:64]=global_addr, [127:126]=2
DEVI u32x4 tdm_g0(unsigned lds, unsigned long long ga) {
  u32x4 g;
  g[0] = 1u;
  g[1] = lds;
  g[2] = (unsigned)(ga & 0xFFFFFFFFull);
  g[3] = (unsigned)((ga >> 32) & 0x01FFFFFFull) | (2u << 30);
  return g;
}
// Group 1 for the h tile: data_size=2B, pad_enable, pad_interval=256 DWORDs,
// pad_amount=1 DWORD (row 512 ushorts -> LDS stride 514), tensor_dim0=512,
// tensor_dim1=32, tile 512x32, tensor_dim0_stride=512.
DEVI i32x8 tdm_g1_h() {
  i32x8 g = {};
  g[0] = (1 << 16) | (1 << 20) | (7 << 22);        // data_size|pad_en|interval
  g[1] = (int)((Hq & 0xFFFF) << 16);               // tensor_dim0 lo16
  g[2] = (int)(Bq << 16);                          // dim0 hi=0 | tensor_dim1 lo16
  g[3] = (int)(Hq << 16);                          // dim1 hi=0 | tile_dim0
  g[4] = Bq;                                       // tile_dim1 | tile_dim2=0
  g[5] = Hq;                                       // tensor_dim0_stride lo32
  g[6] = 0; g[7] = 0;
  return g;
}

// ---------------------------------------------------------------------------
__global__ void k_zero(unsigned int* sync) {
  if (threadIdx.x < 64) sync[threadIdx.x] = 0u;
}

// Embedding gather: ids[8192] -> bf16 x[8192][320] (cols 300..319 zero).
__global__ void k_embed(const int* __restrict__ ids, const float* __restrict__ emb,
                        unsigned short* __restrict__ x) {
  int tok = blockIdx.x;
  int row = ids[tok];
  const float* src = emb + (size_t)row * Eq;
  for (int e = threadIdx.x; e < EP; e += blockDim.x)
    x[(size_t)tok * EP + e] = (e < Eq) ? f2bf(src[e]) : (unsigned short)0;
}

// Pack W[srcN][srcK] (f32, row-major) into WMMA B-fragments of W^T (K x N):
// dst[((kt*(Npad/16)+nt)*32+lane)*16+e] = W[nt*16+lane%16][kt*32+kmap(e,lane/16)]
__global__ void k_pack(const float* __restrict__ W, unsigned short* __restrict__ dst,
                       int Npad, int Kpad, int srcN, int srcK) {
  int idx = blockIdx.x * blockDim.x + threadIdx.x;
  int total = Kpad * Npad;
  if (idx >= total) return;
  int e    = idx & 15;
  int lane = (idx >> 4) & 31;
  int rest = idx >> 9;
  int ntiles = Npad >> 4;
  int nt = rest % ntiles;
  int kt = rest / ntiles;
  int n = nt * 16 + (lane & 15);
  int k = kt * 32 + kmap(e, lane >> 4);
  float v = (n < srcN && k < srcK) ? W[(size_t)n * srcK + k] : 0.f;
  dst[idx] = f2bf(v);
}

// Reverse each sequence within its valid length (row copy, lda bf16 cols).
__global__ void k_reverse(const unsigned short* __restrict__ x,
                          unsigned short* __restrict__ xr,
                          const int* __restrict__ length, int lda) {
  int tok = blockIdx.x; int b = tok / Tq; int t = tok % Tq;
  int len = length[b];
  int idx = (t < len) ? (len - 1 - t) : t;
  const unsigned short* s = x + (size_t)(b * Tq + idx) * lda;
  unsigned short* d = xr + (size_t)tok * lda;
  for (int k = threadIdx.x; k < lda; k += blockDim.x) d[k] = s[k];
}

// ---------------------------------------------------------------------------
// Generic WMMA GEMM: D[M][N] = A[M][Kpad](bf16,row-major) * Bfrag + bias
// 256 threads, block tile MB=MT*16 x NB=NT*16, A staged through padded LDS.
// EPI: 0 f32 store (ldo) | 1 relu->bf16 store (ldo) | 2 f32 store, n<17 only
//      3 time-major pre store: row=b*T+t -> out[(t*B+b)*G4 + n]
template<int MT, int NT, int EPI>
__global__ void k_gemm(const unsigned short* __restrict__ A, int lda, int M,
                       const unsigned short* __restrict__ Bf, int N, int Kpad,
                       const float* __restrict__ bias,
                       float* __restrict__ outF, unsigned short* __restrict__ outH,
                       int ldo) {
  constexpr int MB = MT * 16, NB = NT * 16;
  constexpr int TPW = (MT * NT) / 8;
  __shared__ unsigned short As[MB][34];                 // +2 pad: conflict-free
  const int gridM = M / MB;
  const int bm = blockIdx.x % gridM, bn = blockIdx.x / gridM;
  const int tid = threadIdx.x, wv = tid >> 5, lane = tid & 31;
  const int half = lane >> 4, l16 = lane & 15;
  const int ntilesN = N >> 4;

  f32x8 acc[TPW];
#pragma unroll
  for (int i = 0; i < TPW; i++) acc[i] = {};

  const int KT = Kpad / 32;
  for (int kb = 0; kb < KT; kb++) {
    __syncthreads();
    {   // stage A block (MB x 32 bf16), 16B per thread
      U16 u;
      int r = tid >> 2, c = (tid & 3) * 8;
      u.q = *(const uint4*)(A + (size_t)(bm * MB + r) * lda + kb * 32 + c);
#pragma unroll
      for (int q = 0; q < 8; q++) As[r][c + q] = u.s[q];
    }
    __syncthreads();
#pragma unroll
    for (int i = 0; i < TPW; i++) {
      int tI = wv * TPW + i;
      int mt = tI % MT, nt = tI / MT;
      BFrag a;
#pragma unroll
      for (int e = 0; e < 16; e++)
        a.s[e] = As[mt * 16 + l16][kmap(e, half)];
      BFrag b;
      const uint4* bp = (const uint4*)
          (Bf + ((size_t)(kb * ntilesN + bn * NT + nt) * 32 + lane) * 16);
      b.q[0] = bp[0]; b.q[1] = bp[1];
      if (kb + 1 < KT)   // stream-ahead: weight frags are touched exactly once
        __builtin_prefetch(
            Bf + ((size_t)((kb + 1) * ntilesN + bn * NT + nt) * 32 + lane) * 16,
            0, 1);
      acc[i] = wmma_bf16(a.v, b.v, acc[i]);
    }
  }

#pragma unroll
  for (int i = 0; i < TPW; i++) {
    int tI = wv * TPW + i;
    int mt = tI % MT, nt = tI / MT;
    int n = bn * NB + nt * 16 + l16;
    float bv = 0.f;
    if (EPI == 2) { if (n < Cq) bv = bias[n]; } else bv = bias[n];
#pragma unroll
    for (int r = 0; r < 8; r++) {
      int m = bm * MB + mt * 16 + r + 8 * half;
      float v = acc[i][r] + bv;
      if (EPI == 0) {
        outF[(size_t)m * ldo + n] = v;
      } else if (EPI == 1) {
        outH[(size_t)m * ldo + n] = f2bf(fmaxf(v, 0.f));
      } else if (EPI == 2) {
        if (n < Cq) outF[(size_t)m * Cq + n] = v;
      } else {                       // EPI==3: [b*T+t] -> [t*B+b]
        int bb = m >> 8, tt = m & 255;
        outF[((size_t)tt * Bq + bb) * G4 + n] = v;
      }
    }
  }
}

// ---------------------------------------------------------------------------
// Persistent LSTM: pre[T][B][4H] f32, h ping-pong bf16 [32][512] in global,
// cell state in LDS. 16 WGs, WG j owns h columns [32j, 32j+32).
// h staged into LDS each step with one TENSOR_LOAD_TO_LDS (wave 0 issues).
__global__ void k_lstm(const float* __restrict__ pre,
                       unsigned short* __restrict__ hA,
                       unsigned short* __restrict__ hB,
                       const int* __restrict__ length,
                       unsigned short* __restrict__ y, int ldy, int ycol0, int rev,
                       const unsigned short* __restrict__ WhhF,
                       unsigned int* __restrict__ sync) {
  __shared__ unsigned short Hs[32][514];   // 514 = 512 + TDM pad (1 DWORD/row)
  __shared__ float Cs[32][33];
  __shared__ float Gs[4][32][33];
  const int j = blockIdx.x;
  const int tid = threadIdx.x, wv = tid >> 5, lane = tid & 31;
  const int half = lane >> 4, l16 = lane & 15;
  const int hcol0 = j * 32;
  const int g = wv >> 1;                         // gate group (i,f,g,o)
  const int ntG = g * 32 + j * 2 + (wv & 1);     // gate N-tile of 128

  // TDM descriptors for the two ping-pong h buffers (generic LDS address:
  // low 32 bits of a shared-aperture pointer are the wave LDS byte offset).
  const unsigned ldsHs = (unsigned)(size_t)(&Hs[0][0]);
  const u32x4 g0A = tdm_g0(ldsHs, (unsigned long long)hA);
  const u32x4 g0B = tdm_g0(ldsHs, (unsigned long long)hB);
  const i32x8 g1  = tdm_g1_h();
  const i32x4 gz  = {};
  const i32x8 gz8 = {};

  for (int q = tid; q < 32 * 32; q += 256) {     // zero c and h columns
    int m = q >> 5, cl = q & 31;
    Cs[m][cl] = 0.f;
    hA[m * Hq + hcol0 + cl] = 0;
  }
  gsync(sync, NWG, 1);

  for (int t = 0; t < Tq; t++) {
    unsigned short* hnxt = (t & 1) ? hA : hB;
    if (tid < 32) {                 // wave 0: DMA h tile (32x512 bf16) to LDS
      __builtin_amdgcn_tensor_load_to_lds((t & 1) ? g0B : g0A, g1,
                                          gz, gz, gz8, 0);
      __builtin_amdgcn_s_wait_tensorcnt(0);
    }
    __syncthreads();

    f32x8 acc0 = {}, acc1 = {};
#pragma unroll 4
    for (int kb = 0; kb < Hq / 32; kb++) {
      BFrag b;
      const uint4* bp = (const uint4*)
          (WhhF + ((size_t)(kb * (G4 / 16) + ntG) * 32 + lane) * 16);
      b.q[0] = bp[0]; b.q[1] = bp[1];
      if (kb + 1 < Hq / 32)
        __builtin_prefetch(
            WhhF + ((size_t)((kb + 1) * (G4 / 16) + ntG) * 32 + lane) * 16,
            0, 1);
      BFrag a0, a1;
#pragma unroll
      for (int e = 0; e < 16; e++) {
        int k = kb * 32 + kmap(e, half);
        a0.s[e] = Hs[l16][k];
        a1.s[e] = Hs[16 + l16][k];
      }
      acc0 = wmma_bf16(a0.v, b.v, acc0);
      acc1 = wmma_bf16(a1.v, b.v, acc1);
    }

    {   // gates = recurrent + pre, exchange through LDS
      int nl = (wv & 1) * 16 + l16;
      int gcol = g * Hq + hcol0 + nl;
      const float* pt = pre + (size_t)t * Bq * G4;
#pragma unroll
      for (int r = 0; r < 8; r++) {
        int m0 = r + 8 * half;
        Gs[g][m0][nl]      = acc0[r] + pt[(size_t)m0 * G4 + gcol];
        Gs[g][m0 + 16][nl] = acc1[r] + pt[(size_t)(m0 + 16) * G4 + gcol];
      }
    }
    __syncthreads();

    for (int q = tid; q < 1024; q += 256) {      // element-wise cell update
      int m = q >> 5, cl = q & 31;
      int len = length[m];
      float mk = (t < len) ? 1.f : 0.f;
      float gi = Gs[0][m][cl], gf = Gs[1][m][cl];
      float gg = Gs[2][m][cl], go = Gs[3][m][cl];
      float c_old = Cs[m][cl];
      float c_new = sigf(gf) * c_old + sigf(gi) * tanhf(gg);
      float h_new = sigf(go) * tanhf(c_new);
      Cs[m][cl] = mk * c_new + (1.f - mk) * c_old;
      float h_prev = bf2f(Hs[m][hcol0 + cl]);
      float h_keep = mk * h_new + (1.f - mk) * h_prev;
      hnxt[m * Hq + hcol0 + cl] = f2bf(h_keep);
      int tout = t;
      if (rev) tout = (t < len) ? (len - 1 - t) : t;
      y[((size_t)m * Tq + tout) * ldy + ycol0 + cl] = f2bf(mk * h_new);
    }
    gsync(sync, NWG, t + 2);
  }
}

// ---------------------------------------------------------------------------
// CRF NLL: one block, one wave per batch row, lane=tag. out0[0] = nll.
__global__ void k_crf(const float* __restrict__ em,    // [B][T][17]
                      const int* __restrict__ tags,
                      const int* __restrict__ length,
                      const float* __restrict__ cs, const float* __restrict__ ce,
                      const float* __restrict__ ctr,
                      float* __restrict__ out0) {
  __shared__ float trs[Cq * Cq], sts[Cq], ens[Cq], llh[Bq];
  int tid = threadIdx.x;
  for (int i = tid; i < Cq * Cq; i += blockDim.x) trs[i] = ctr[i];
  if (tid < Cq) { sts[tid] = cs[tid]; ens[tid] = ce[tid]; }
  __syncthreads();
  int wv = tid >> 5, lane = tid & 31, b = wv;
  int len = length[b];
  const float* eb = em + (size_t)b * Tq * Cq;
  int c = (lane < Cq) ? lane : 0;
  float alpha = (lane < Cq) ? sts[lane] + eb[lane] : -1e30f;
  for (int t = 1; t < Tq; t++) {
    bool mt = (t < len);
    float et = (lane < Cq) ? eb[t * Cq + lane] : 0.f;
    float m2 = -1e30f;
    for (int k = 0; k < Cq; k++)
      m2 = fmaxf(m2, __shfl(alpha, k, 32) + trs[k * Cq + c]);
    float s = 0.f;
    for (int k = 0; k < Cq; k++)
      s += __expf(__shfl(alpha, k, 32) + trs[k * Cq + c] - m2);
    float nxt = __logf(s) + m2 + et;
    if (mt && lane < Cq) alpha = nxt;
  }
  float v = (lane < Cq) ? alpha + ens[lane] : -1e30f;
  float mx = v;
  for (int o = 16; o; o >>= 1) mx = fmaxf(mx, __shfl_xor(mx, o, 32));
  float s = (lane < Cq) ? __expf(v - mx) : 0.f;
  for (int o = 16; o; o >>= 1) s += __shfl_xor(s, o, 32);
  float logZ = __logf(s) + mx;
  if (lane == 0) {
    const int* tg = tags + b * Tq;
    int prev = tg[0];
    float sc = sts[prev] + eb[prev];
    for (int t = 1; t < Tq; t++) {
      if (t < len) {
        int cur = tg[t];
        sc += trs[prev * Cq + cur] + eb[t * Cq + cur];
        prev = cur;
      }
    }
    sc += ens[prev];
    llh[b] = sc - logZ;
  }
  __syncthreads();
  if (tid == 0) {
    float acc = 0.f;
    for (int i = 0; i < Bq; i++) acc += llh[i];
    out0[0] = -acc / (float)Bq;
  }
}

// ---------------------------------------------------------------------------
extern "C" void kernel_launch(void* const* d_in, const int* in_sizes, int n_in,
                              void* d_out, int out_size, void* d_ws, size_t ws_size,
                              hipStream_t stream) {
  const int*   ids    = (const int*)  d_in[0];
  const int*   tags   = (const int*)  d_in[1];
  const int*   length = (const int*)  d_in[2];
  // d_in[3] = mask (recomputed from length on device)
  const float* emb    = (const float*)d_in[4];
  const float* Wih[2][2] = {{(const float*)d_in[5],  (const float*)d_in[8]},
                            {(const float*)d_in[11], (const float*)d_in[14]}};
  const float* Whh[2][2] = {{(const float*)d_in[6],  (const float*)d_in[9]},
                            {(const float*)d_in[12], (const float*)d_in[15]}};
  const float* bb [2][2] = {{(const float*)d_in[7],  (const float*)d_in[10]},
                            {(const float*)d_in[13], (const float*)d_in[16]}};
  const float* W1 = (const float*)d_in[17];
  const float* b1 = (const float*)d_in[18];
  const float* W2 = (const float*)d_in[19];
  const float* b2 = (const float*)d_in[20];
  const float* crf_s = (const float*)d_in[21];
  const float* crf_e = (const float*)d_in[22];
  const float* crf_t = (const float*)d_in[23];
  float* out = (float*)d_out;                       // [0]=nll, [1..]=emissions

  // workspace carve-out (256B aligned)
  size_t o = 0;
  auto take = [&](size_t bytes) -> size_t {
    size_t r = o; o = (o + bytes + 255) & ~(size_t)255; return r;
  };
  char* ws = (char*)d_ws;
  unsigned int*   sync = (unsigned int*)  (ws + take(256));
  unsigned short* hA   = (unsigned short*)(ws + take((size_t)Bq * Hq * 2));
  unsigned short* hB   = (unsigned short*)(ws + take((size_t)Bq * Hq * 2));
  unsigned short* buf0 = (unsigned short*)(ws + take((size_t)Bq * Tq * H2 * 2));
  unsigned short* buf1 = (unsigned short*)(ws + take((size_t)Bq * Tq * H2 * 2));
  unsigned short* xr   = (unsigned short*)(ws + take((size_t)Bq * Tq * H2 * 2));
  float*          pre  = (float*)         (ws + take((size_t)Tq * Bq * G4 * 4));
  unsigned short* h1   = (unsigned short*)(ws + take((size_t)Bq * Tq * Hq * 2));
  unsigned short* wihF[2][2], *whhF[2][2];
  const int kin[2] = {EP, H2};
  for (int l = 0; l < 2; l++)
    for (int d = 0; d < 2; d++) {
      wihF[l][d] = (unsigned short*)(ws + take((size_t)kin[l] * G4 * 2));
      whhF[l][d] = (unsigned short*)(ws + take((size_t)Hq * G4 * 2));
    }
  unsigned short* w1F = (unsigned short*)(ws + take((size_t)H2 * Hq * 2));
  unsigned short* w2F = (unsigned short*)(ws + take((size_t)Hq * 32 * 2));

  k_zero<<<1, 64, 0, stream>>>(sync);
  k_embed<<<Bq * Tq, EP, 0, stream>>>(ids, emb, buf0);

  auto pack = [&](const float* W, unsigned short* dst, int Np, int Kp,
                  int sN, int sK) {
    int total = Kp * Np;
    k_pack<<<(total + 255) / 256, 256, 0, stream>>>(W, dst, Np, Kp, sN, sK);
  };
  for (int l = 0; l < 2; l++)
    for (int d = 0; d < 2; d++) {
      pack(Wih[l][d], wihF[l][d], G4, kin[l], G4, l == 0 ? Eq : H2);
      pack(Whh[l][d], whhF[l][d], G4, Hq, G4, Hq);
    }
  pack(W1, w1F, Hq, H2, Hq, H2);
  pack(W2, w2F, 32, Hq, Cq, Hq);

  const int M = Bq * Tq;
  unsigned short* xin  = buf0;
  unsigned short* xout = buf1;
  for (int l = 0; l < 2; l++) {
    int K = kin[l];
    int gPre = (M / 64) * (G4 / 64);
    // forward direction
    k_gemm<4, 4, 3><<<gPre, 256, 0, stream>>>(xin, K, M, wihF[l][0], G4, K,
                                              bb[l][0], pre, nullptr, G4);
    k_lstm<<<NWG, 256, 0, stream>>>(pre, hA, hB, length, xout, H2, 0, 0,
                                    whhF[l][0], sync + 2 * l);
    // backward direction: reverse input, run, write reversed
    k_reverse<<<M, 256, 0, stream>>>(xin, xr, length, K);
    k_gemm<4, 4, 3><<<gPre, 256, 0, stream>>>(xr, K, M, wihF[l][1], G4, K,
                                              bb[l][1], pre, nullptr, G4);
    k_lstm<<<NWG, 256, 0, stream>>>(pre, hA, hB, length, xout, H2, Hq, 1,
                                    whhF[l][1], sync + 2 * l + 1);
    unsigned short* t = xin; xin = xout; xout = t;   // xin now holds [B,T,2H]
  }

  // FFN1: relu(out @ W1^T + b1) -> h1 bf16 [8192][512]
  k_gemm<4, 4, 1><<<(M / 64) * (Hq / 64), 256, 0, stream>>>(
      xin, H2, M, w1F, Hq, H2, b1, nullptr, h1, Hq);
  // FFN2: emissions f32 [8192][17] -> d_out+1
  k_gemm<4, 2, 2><<<(M / 64) * (32 / 32), 256, 0, stream>>>(
      h1, Hq, M, w2F, 32, Hq, b2, out + 1, nullptr, Cq);
  // CRF NLL -> d_out[0]
  k_crf<<<1, 1024, 0, stream>>>(out + 1, tags, length, crf_s, crf_e, crf_t, out);
}
